// ProgrammedTileFFN_63806034149898
// MI455X (gfx1250) — matrix operand order repaired and checked by reading, not compiled, not run on hardware
//
#include <hip/hip_runtime.h>
#include <hip/hip_bf16.h>
#include <math.h>
#include <stdint.h>

// ---------------------------------------------------------------------------
// ProgrammedTileFFN for MI455X (gfx1250, wave32, WMMA + TDM, double-buffered).
// Pipeline:
//   0. zero per-tile counters
//   1. convert x -> bf16 (row-major); convert+transpose all weights -> bf16 [N][K]
//   2. router GEMM1 (WMMA bf16) -> Hr
//   3. router logits/argmax + per-tile token lists (atomics)
//   4. routed tile GEMM1 + GELU (gathered A rows, WMMA)
//   5. routed tile GEMM2        (gathered A rows, WMMA)
//   6. output projection (WMMA, f32 out)
// GEMM staging: 2-deep LDS double buffer; stage i+1 (TDM tensor_load_to_lds
// for B/[A], global_load_b128->regs for gathered A) overlaps WMMA of stage i;
// s_wait_tensorcnt + one barrier per iteration.
// ---------------------------------------------------------------------------

typedef __attribute__((ext_vector_type(16))) __bf16 v16bf;
typedef __attribute__((ext_vector_type(8)))  float  v8f;
typedef unsigned int u32x4 __attribute__((ext_vector_type(4)));
typedef int          i32x4 __attribute__((ext_vector_type(4)));
typedef int          i32x8 __attribute__((ext_vector_type(8)));

#define D_MODEL   1024
#define NUM_TILES 8
#define BATCH     4096

#define BLK_M 128
#define BLK_N 128
#define BLK_K 32
#define A_ST  40   // LDS K-stride in ushorts: 64B row + 16B pad = 80B (16B-aligned)
#define B_ST  40

#if defined(__has_builtin)
#  if __has_builtin(__builtin_amdgcn_tensor_load_to_lds) && \
      __has_builtin(__builtin_amdgcn_s_wait_tensorcnt)
#    define HAVE_TDM 1
#  endif
#endif
#ifndef HAVE_TDM
#  define HAVE_TDM 0
#endif

static __device__ __forceinline__ unsigned short f32_to_bf16(float f) {
  unsigned u = __builtin_bit_cast(unsigned, f);
  u += 0x7FFFu + ((u >> 16) & 1u);              // round-to-nearest-even
  return (unsigned short)(u >> 16);
}
static __device__ __forceinline__ unsigned pack_bf16x2(float a, float b) {
  unsigned ua = __builtin_bit_cast(unsigned, a);
  unsigned ub = __builtin_bit_cast(unsigned, b);
  ua += 0x7FFFu + ((ua >> 16) & 1u);
  ub += 0x7FFFu + ((ub >> 16) & 1u);
  return (ua >> 16) | (ub & 0xFFFF0000u);
}
static __device__ __forceinline__ float bf16_to_f32(unsigned short h) {
  unsigned u = ((unsigned)h) << 16;
  return __builtin_bit_cast(float, u);
}
static __device__ __forceinline__ float gelu_exact(float x) {
  return 0.5f * x * (1.0f + erff(x * 0.7071067811865475f));
}

union Frag { uint4 q[2]; v16bf v; };

#if HAVE_TDM
// TDM 2D tile load: 128 rows x 32 bf16, row stride K elements, into LDS with
// pad_interval = 16 DWORDs (64B) and pad_amount = 4 DWORDs (16B) -> 80B LDS
// row stride == A_ST/B_ST. D# per CDNA5 ISA §8.3/8.4 (group0 + group1).
static __device__ __forceinline__ void tdm_load_tile_128x32(
    unsigned lds_off, const unsigned short* gsrc, int K) {
  unsigned long long ga = (unsigned long long)(uintptr_t)gsrc;
  u32x4 g0;
  g0[0] = 1u;                                        // count=1, user desc
  g0[1] = lds_off;                                   // lds_addr[31:0]
  g0[2] = (unsigned)ga;                              // global_addr[31:0]
  g0[3] = (unsigned)((ga >> 32) & 0x1FFFFFFull)      // global_addr[56:32]
          | (2u << 30);                              // type = 2 ("image")
  i32x8 g1;
  g1[0] = (int)((1u << 16)        // data_size = 2 bytes
              | (1u << 20)        // pad_enable
              | (3u << 22)        // pad_interval: 16 DWORDs
              | (3u << 25));      // pad_amount:   4 DWORDs
  g1[1] = (int)(((unsigned)K & 0xFFFFu) << 16);              // tensor_dim0 lo
  g1[2] = (int)((((unsigned)K >> 16) & 0xFFFFu)              // tensor_dim0 hi
              | (128u << 16));                               // tensor_dim1 lo
  g1[3] = (int)(32u << 16);                                  // tile_dim0 = 32
  g1[4] = 128;                                               // tile_dim1 = 128
  g1[5] = K;                                                 // dim0_stride lo
  g1[6] = 0;
  g1[7] = 0;
  i32x4 z4 = {0, 0, 0, 0};
#if __clang_major__ >= 23
  i32x8 z8 = {0, 0, 0, 0, 0, 0, 0, 0};
  __builtin_amdgcn_tensor_load_to_lds(g0, g1, z4, z4, z8, 0);
#else
  __builtin_amdgcn_tensor_load_to_lds(g0, g1, z4, z4, 0);
#endif
}
#endif

// ---------------------------------------------------------------------------
// bf16 WMMA GEMM:  C[rows] = act(A @ Bt^T + bias)
//   A : bf16 row-major [*, K] (rows gathered via token list if GATHER)
//   Bt: bf16 TRANSPOSED [N, K] (per-tile stride), rows land in LDS unmodified
// Block tile 128x128x32, 256 threads = 8 waves, wave tile 32x64 (2x4 WMMA).
// 2-deep LDS double buffer: stage i+1 loads overlap stage i WMMA.
// ---------------------------------------------------------------------------
template<bool GATHER, bool DO_GELU, bool OUT_BF16>
__global__ __launch_bounds__(256)
void wmma_gemm_kernel(const unsigned short* __restrict__ Ag,
                      const unsigned short* __restrict__ Btg,
                      const float* __restrict__ biasg,
                      void* __restrict__ Cg,
                      int K, int N,
                      const int* __restrict__ counts,
                      const int* __restrict__ lists,
                      long long b_tile_stride, int bias_tile_stride)
{
  __shared__ unsigned short lds_a[2 * BLK_M * A_ST];   // A tiles [buf][m][k]
  __shared__ unsigned short lds_b[2 * BLK_N * B_ST];   // B tiles [buf][n][k]

  const int tile = blockIdx.z;
  int m_count = BATCH;
  const int* list = nullptr;
  if (GATHER) {
    m_count = counts[tile];
    if ((int)blockIdx.y * BLK_M >= m_count) return;   // handles count==0 too
    list = lists + tile * BATCH;
  }
  const unsigned short* Bt = Btg + (long long)tile * b_tile_stride;
  const float* bias = biasg + tile * bias_tile_stride;

  const int tid  = threadIdx.x;
  const int wave = tid >> 5;
  const int lane = tid & 31;
  const int wm   = (wave >> 1) * 32;
  const int wn   = (wave &  1) * 64;
  const int lh   = lane >> 4;
  const int l16  = lane & 15;
  const int block_m = blockIdx.y * BLK_M;
  const int block_n = blockIdx.x * BLK_N;

  v8f acc[2][4];
  const v8f vz = {0.f,0.f,0.f,0.f,0.f,0.f,0.f,0.f};
#pragma unroll
  for (int i = 0; i < 2; i++)
#pragma unroll
    for (int j = 0; j < 4; j++) acc[i][j] = vz;

  // per-thread staging coords: thread t copies 32B = 16 bf16 of row m, half h
  const int sm = tid >> 1;
  const int sh = (tid & 1) * 16;

  // hoist gathered/fixed A source row pointer (k-independent)
  const unsigned short* arow;
  if (GATHER) {
    int r  = block_m + sm;
    int rc = r < m_count ? r : m_count - 1;   // clamp: duplicate rows benign
    arow = Ag + (size_t)list[rc] * K + sh;
  } else {
    arow = Ag + (size_t)(block_m + sm) * K + sh;
  }
#if !HAVE_TDM
  const unsigned short* brow = Bt + (size_t)(block_n + sm) * K + sh;
  uint4 rb0, rb1;                     // B stage registers (fallback path)
#endif
  uint4 ra0, ra1;                     // A stage registers (gather / fallback)

  // ---- stage issue: start global->LDS/regs transfers for k0 into buf ----
  auto issue = [&](int k0, int buf) {
#if HAVE_TDM
    if (wave == 0) {
      tdm_load_tile_128x32(
          (unsigned)(uintptr_t)(void*)(lds_b + buf * (BLK_N * B_ST)),
          Bt + (size_t)block_n * K + k0, K);
      if (!GATHER)
        tdm_load_tile_128x32(
            (unsigned)(uintptr_t)(void*)(lds_a + buf * (BLK_M * A_ST)),
            Ag + (size_t)block_m * K + k0, K);
    }
    if (GATHER) {
      const uint4* s = (const uint4*)(arow + k0);
      ra0 = s[0]; ra1 = s[1];
    }
#else
    { const uint4* s = (const uint4*)(brow + k0); rb0 = s[0]; rb1 = s[1]; }
    { const uint4* s = (const uint4*)(arow + k0); ra0 = s[0]; ra1 = s[1]; }
#endif
  };
  // ---- stage commit: register->LDS stores (after compute of prior stage) ----
  auto commit = [&](int buf) {
#if HAVE_TDM
    if (GATHER) {
      unsigned short* la = lds_a + buf * (BLK_M * A_ST);
      *(uint4*)&la[sm * A_ST + sh]     = ra0;
      *(uint4*)&la[sm * A_ST + sh + 8] = ra1;
    }
#else
    unsigned short* la = lds_a + buf * (BLK_M * A_ST);
    unsigned short* lb = lds_b + buf * (BLK_N * B_ST);
    *(uint4*)&lb[sm * B_ST + sh]     = rb0;
    *(uint4*)&lb[sm * B_ST + sh + 8] = rb1;
    *(uint4*)&la[sm * A_ST + sh]     = ra0;
    *(uint4*)&la[sm * A_ST + sh + 8] = ra1;
#endif
  };
  // ---- compute: fragments per ISA 16-bit WMMA layouts + 8 WMMAs ----
  auto compute = [&](int buf) {
    const unsigned short* la = lds_a + buf * (BLK_M * A_ST);
    const unsigned short* lb = lds_b + buf * (BLK_N * B_ST);
    Frag af[2], bfr[4];
    const int ak = lh * 8;
#pragma unroll
    for (int mt = 0; mt < 2; mt++) {
      const unsigned short* p = &la[(wm + mt * 16 + l16) * A_ST + ak];
      af[mt].q[0] = *(const uint4*)(p);
      af[mt].q[1] = *(const uint4*)(p + 16);
    }
    const int bk = lh * 8;
#pragma unroll
    for (int nt = 0; nt < 4; nt++) {
      const unsigned short* p = &lb[(wn + nt * 16 + l16) * B_ST + bk];
      bfr[nt].q[0] = *(const uint4*)(p);
      bfr[nt].q[1] = *(const uint4*)(p + 16);
    }
#pragma unroll
    for (int mt = 0; mt < 2; mt++)
#pragma unroll
      for (int nt = 0; nt < 4; nt++)
        acc[mt][nt] = __builtin_amdgcn_wmma_f32_16x16x32_bf16(
            false, af[mt].v, false, bfr[nt].v, (short)0, acc[mt][nt], false, false);
  };

  // ---- software pipeline: prologue stages k=0, loop overlaps i+1 with i ----
  const int nk = K / BLK_K;
  issue(0, 0);
  commit(0);
#if HAVE_TDM
  if (wave == 0) __builtin_amdgcn_s_wait_tensorcnt(0);
#endif
  __syncthreads();                               // buf0 ready

  for (int i = 0; i < nk; i++) {
    const int cur = i & 1;
    const bool more = (i + 1) < nk;
    if (more) issue((i + 1) * BLK_K, 1 - cur);   // overlaps compute below
    compute(cur);
    if (more) {
      commit(1 - cur);                           // reg->LDS after compute
#if HAVE_TDM
      if (wave == 0) __builtin_amdgcn_s_wait_tensorcnt(0);
#endif
    }
    __syncthreads();                             // next buffer ready/free
  }

  // ---- epilogue: bias (+GELU), scatter rows back by token id ----
#pragma unroll
  for (int mt = 0; mt < 2; mt++) {
#pragma unroll
    for (int nt = 0; nt < 4; nt++) {
#pragma unroll
      for (int r = 0; r < 8; r++) {
        int row = block_m + wm + mt * 16 + lh * 8 + r;
        int col = block_n + wn + nt * 16 + l16;
        if (GATHER && row >= m_count) continue;
        float v = acc[mt][nt][r] + bias[col];
        if (DO_GELU) v = gelu_exact(v);
        int orow = GATHER ? list[row] : row;
        if (OUT_BF16) ((unsigned short*)Cg)[(size_t)orow * N + col] = f32_to_bf16(v);
        else          ((float*)Cg)[(size_t)orow * N + col] = v;
      }
    }
  }
}

// ---------------------------------------------------------------------------
// f32 -> bf16 row-major convert (4 elems/thread, packed)
// ---------------------------------------------------------------------------
__global__ __launch_bounds__(256)
void cvt_bf16_kernel(const float* __restrict__ in, unsigned short* __restrict__ out) {
  size_t i = ((size_t)blockIdx.x * 256 + threadIdx.x) * 4;
  float4 v = *(const float4*)(in + i);
  uint2 p;
  p.x = pack_bf16x2(v.x, v.y);
  p.y = pack_bf16x2(v.z, v.w);
  *(uint2*)(out + i) = p;
}

// ---------------------------------------------------------------------------
// f32 [K][N] -> bf16 [N][K] transpose-convert via 32x32 LDS tile
// ---------------------------------------------------------------------------
__global__ __launch_bounds__(256)
void transpose_cvt_kernel(const float* __restrict__ in, unsigned short* __restrict__ out,
                          int K, int N, long long in_tile_stride, long long out_tile_stride)
{
  __shared__ unsigned short t[32][36];   // pad: 72B row stride (8B aligned)
  const float* inp = in + (long long)blockIdx.z * in_tile_stride;
  unsigned short* outp = out + (long long)blockIdx.z * out_tile_stride;
  const int n0 = blockIdx.x * 32;
  const int k0 = blockIdx.y * 32;
  const int r = threadIdx.x >> 3;          // 0..31
  const int c = (threadIdx.x & 7) * 4;     // 0..28
  float4 v = *(const float4*)(inp + (size_t)(k0 + r) * N + n0 + c);
  t[c + 0][r] = f32_to_bf16(v.x);
  t[c + 1][r] = f32_to_bf16(v.y);
  t[c + 2][r] = f32_to_bf16(v.z);
  t[c + 3][r] = f32_to_bf16(v.w);
  __syncthreads();
  *(uint2*)(outp + (size_t)(n0 + r) * K + k0 + c) = *(const uint2*)&t[r][c];
}

// ---------------------------------------------------------------------------
// Router tail: logits = (Hr @ Wr2 + br2)/T, argmax, build per-tile token lists
// ---------------------------------------------------------------------------
__global__ __launch_bounds__(256)
void router_kernel(const unsigned short* __restrict__ Hr,
                   const float* __restrict__ Wr2,
                   const float* __restrict__ br2,
                   float* __restrict__ logits,
                   float* __restrict__ tidx_f,
                   int*   __restrict__ tidx,
                   int*   __restrict__ counts,
                   int*   __restrict__ lists)
{
  int token = blockIdx.x * 256 + threadIdx.x;
  float a[NUM_TILES];
#pragma unroll
  for (int j = 0; j < NUM_TILES; j++) a[j] = 0.f;
  const unsigned short* hrow = Hr + (size_t)token * D_MODEL;
  for (int k = 0; k < D_MODEL; k += 2) {
    float h0 = bf16_to_f32(hrow[k]);
    float h1 = bf16_to_f32(hrow[k + 1]);
#pragma unroll
    for (int j = 0; j < NUM_TILES; j++)
      a[j] += h0 * Wr2[k * NUM_TILES + j] + h1 * Wr2[(k + 1) * NUM_TILES + j];
  }
  int best = 0; float bv = -3.4e38f;
#pragma unroll
  for (int j = 0; j < NUM_TILES; j++) {
    float l = (a[j] + br2[j]) * 2.0f;   // / TEMPERATURE (0.5)
    logits[(size_t)token * NUM_TILES + j] = l;
    if (l > bv) { bv = l; best = j; }
  }
  tidx_f[token] = (float)best;
  tidx[token]   = best;
  int pos = atomicAdd(&counts[best], 1);
  lists[best * BATCH + pos] = token;
}

__global__ void init_counts_kernel(int* counts) {
  if (threadIdx.x < NUM_TILES) counts[threadIdx.x] = 0;
}

// ---------------------------------------------------------------------------
extern "C" void kernel_launch(void* const* d_in, const int* in_sizes, int n_in,
                              void* d_out, int out_size, void* d_ws, size_t ws_size,
                              hipStream_t stream) {
  (void)in_sizes; (void)n_in; (void)out_size; (void)ws_size;
  const float* x   = (const float*)d_in[0];
  const float* Wr1 = (const float*)d_in[1];
  const float* br1 = (const float*)d_in[2];
  const float* Wr2 = (const float*)d_in[3];
  const float* br2 = (const float*)d_in[4];
  const float* W1  = (const float*)d_in[5];
  const float* b1  = (const float*)d_in[6];
  const float* W2  = (const float*)d_in[7];
  const float* b2  = (const float*)d_in[8];
  const float* Wo  = (const float*)d_in[9];
  const float* bo  = (const float*)d_in[10];

  const int D = D_MODEL;
  // d_out = [output(4096x1024) | tile_idx(4096) | route_logits(4096x8)] as f32
  float* out    = (float*)d_out;
  float* tidx_f = out + (size_t)BATCH * D;
  float* logits = tidx_f + BATCH;

  // workspace layout (bf16 activations + bf16 transposed weights)
  unsigned short* xb   = (unsigned short*)d_ws;                    // 4096x1024
  unsigned short* Hr   = xb   + (size_t)BATCH * D;                 // 4096x1024
  unsigned short* H    = Hr   + (size_t)BATCH * D;                 // 4096x2048
  unsigned short* sel  = H    + (size_t)BATCH * 2 * D;             // 4096x1024
  unsigned short* Wr1t = sel  + (size_t)BATCH * D;                 // [D][D]
  unsigned short* W1t  = Wr1t + (size_t)D * D;                     // [T][2D][D]
  unsigned short* W2t  = W1t  + (size_t)NUM_TILES * D * 2 * D;     // [T][D][2D]
  unsigned short* Wot  = W2t  + (size_t)NUM_TILES * 2 * D * D;     // [D][D]
  int* tidx   = (int*)(Wot + (size_t)D * D);
  int* counts = tidx + BATCH;
  int* lists  = counts + 16;                                       // 8 x 4096

  init_counts_kernel<<<1, 32, 0, stream>>>(counts);

  // --- precision/layout prep: bf16 x, bf16 transposed weights ---
  cvt_bf16_kernel<<<(BATCH * D / 4) / 256, 256, 0, stream>>>(x, xb);
  transpose_cvt_kernel<<<dim3(D / 32, D / 32, 1), 256, 0, stream>>>(
      Wr1, Wr1t, D, D, 0, 0);
  transpose_cvt_kernel<<<dim3(2 * D / 32, D / 32, NUM_TILES), 256, 0, stream>>>(
      W1, W1t, D, 2 * D, (long long)D * 2 * D, (long long)D * 2 * D);
  transpose_cvt_kernel<<<dim3(D / 32, 2 * D / 32, NUM_TILES), 256, 0, stream>>>(
      W2, W2t, 2 * D, D, (long long)2 * D * D, (long long)2 * D * D);
  transpose_cvt_kernel<<<dim3(D / 32, D / 32, 1), 256, 0, stream>>>(
      Wo, Wot, D, D, 0, 0);

  // --- Router GEMM1: Hr = gelu(x @ Wr1 + br1) ---
  wmma_gemm_kernel<false, true, true>
    <<<dim3(D / BLK_N, BATCH / BLK_M, 1), 256, 0, stream>>>(
      xb, Wr1t, br1, Hr, D, D, nullptr, nullptr, 0, 0);

  // --- Router logits + argmax + per-tile token lists ---
  router_kernel<<<BATCH / 256, 256, 0, stream>>>(
      Hr, Wr2, br2, logits, tidx_f, tidx, counts, lists);

  // --- Routed tile GEMM1: H[tok] = gelu(x[tok] @ W1[t] + b1[t]) ---
  wmma_gemm_kernel<true, true, true>
    <<<dim3(2 * D / BLK_N, BATCH / BLK_M, NUM_TILES), 256, 0, stream>>>(
      xb, W1t, b1, H, D, 2 * D, counts, lists,
      (long long)D * 2 * D, 2 * D);

  // --- Routed tile GEMM2: sel[tok] = H[tok] @ W2[t] + b2[t] ---
  wmma_gemm_kernel<true, false, true>
    <<<dim3(D / BLK_N, BATCH / BLK_M, NUM_TILES), 256, 0, stream>>>(
      H, W2t, b2, sel, 2 * D, D, counts, lists,
      (long long)2 * D * D, D);

  // --- Output projection: out = sel @ Wo + bo ---
  wmma_gemm_kernel<false, false, false>
    <<<dim3(D / BLK_N, BATCH / BLK_M, 1), 256, 0, stream>>>(
      sel, Wot, bo, out, D, D, nullptr, nullptr, 0, 0);
}
